// ActionModule_53601191854391
// MI455X (gfx1250) — matrix operand order repaired and checked by reading, not compiled, not run on hardware
//
#include <hip/hip_runtime.h>
#include <hip/hip_bf16.h>
#include <stdint.h>

typedef __attribute__((ext_vector_type(16))) __bf16 v16bf;
typedef __attribute__((ext_vector_type(8)))  float  v8f;
typedef __attribute__((ext_vector_type(4)))  int    v4i;

#define B_     2
#define TT_    16
#define S_     1560          // th*tw = 30*52
#define C_     768
#define HEADS_ 24
#define HD_    32
#define M_     49920         // B_*S_*TT_
#define L_     24960         // TT_*S_
#define NF_    61            // (tt-1)*R+1
#define KZP_   800           // 792 padded to 32

// ---- gfx1250 async global->LDS copy (ASYNCcnt path), with safe fallback ----
#define AS1 __attribute__((address_space(1)))
#define AS3 __attribute__((address_space(3)))
#if defined(__has_builtin)
#if __has_builtin(__builtin_amdgcn_global_load_async_to_lds_b128) && \
    __has_builtin(__builtin_amdgcn_s_wait_asynccnt)
#define HAVE_ASYNC_LDS 1
#endif
#endif
#ifndef HAVE_ASYNC_LDS
#define HAVE_ASYNC_LDS 0
#endif

__device__ __forceinline__ void async_cp16(const __bf16* gsrc, __bf16* ldst) {
#if HAVE_ASYNC_LDS
    __builtin_amdgcn_global_load_async_to_lds_b128(
        (AS1 v4i*)(v4i*)(__bf16*)gsrc, (AS3 v4i*)(v4i*)ldst, 0, 0);
#else
    *reinterpret_cast<uint4*>(ldst) = *reinterpret_cast<const uint4*>(gsrc);
#endif
}
__device__ __forceinline__ void async_wait() {
#if HAVE_ASYNC_LDS
    __builtin_amdgcn_s_wait_asynccnt(0);
#endif
}

// ---------------------------------------------------------------------------
// Stage Z = [hs | mouse-history] as bf16, rows r=((bi*S+si)*tt+t), K padded
// ---------------------------------------------------------------------------
__global__ __launch_bounds__(256)
void stage_z_kernel(const float* __restrict__ x, const float* __restrict__ mouse,
                    __bf16* __restrict__ Z)
{
    size_t idx = (size_t)blockIdx.x * 256 + threadIdx.x;
    if (idx >= (size_t)M_ * KZP_) return;
    int c = (int)(idx % KZP_);
    size_t r = idx / KZP_;
    int t  = (int)(r % TT_);
    size_t rs = r / TT_;
    int si = (int)(rs % S_);
    int bi = (int)(rs / S_);
    float v = 0.f;
    if (c < C_) {
        v = x[((size_t)bi * L_ + (size_t)t * S_ + si) * C_ + c];
    } else if (c < 792) {
        int j = c - C_;          // 0..23
        int p = j >> 1, comp = j & 1;
        int mi = 4 * t + p - 12; // R*t + p - PAD_T
        if (mi >= 0) v = mouse[((size_t)bi * NF_ + mi) * 2 + comp];
    }
    Z[idx] = (__bf16)v;
}

// ---------------------------------------------------------------------------
// Convert f32 weight [K x N] -> bf16 [Kp x Np] zero padded
// ---------------------------------------------------------------------------
__global__ __launch_bounds__(256)
void stage_w_kernel(const float* __restrict__ W, __bf16* __restrict__ Ws,
                    int K, int N, int Kp, int Np)
{
    size_t idx = (size_t)blockIdx.x * 256 + threadIdx.x;
    if (idx >= (size_t)Kp * Np) return;
    int n = (int)(idx % Np);
    int k = (int)(idx / Np);
    float v = (k < K && n < N) ? W[(size_t)k * N + n] : 0.f;
    Ws[idx] = (__bf16)v;
}

// ---------------------------------------------------------------------------
// Tiled bf16 WMMA GEMM: C[M x Np] = A[M x Kp] * B[Kp x Np] (+bias, epilogue)
// Block tile 256x128, 8 waves (4 m x 2 n), each wave 64x64 = 4x4 WMMA tiles.
// Double-buffered LDS; A tile via async global->LDS, B staged+transposed.
// EPI: 0 = bias+gelu->bf16 (zero pad cols), 1 = bias->bf16, 2 = ->bf16,
//      3 = residual + row-unpermute -> f32 (final output)
// ---------------------------------------------------------------------------
template<int EPI>
__global__ __launch_bounds__(256)
void gemm_bf16_wmma(const __bf16* __restrict__ A, const __bf16* __restrict__ Bm,
                    const float* __restrict__ bias,
                    __bf16* __restrict__ Cb, float* __restrict__ Cf,
                    const float* __restrict__ resid,
                    int Kp, int Np, int Ntrue)
{
    __shared__ __bf16 As[2][256][40];   // [buf][m][k], padded stride (40 KB)
    __shared__ __bf16 Bs[2][128][40];   // [buf][n][k] transposed     (20 KB)

    const int tid  = threadIdx.x;
    const int wid  = tid >> 5, lane = tid & 31;
    const int half = lane >> 4, l15 = lane & 15;
    const int wm = wid & 3, wn = wid >> 2;
    const int m0 = blockIdx.y * 256;
    const int n0 = blockIdx.x * 128;

    v8f acc[4][4] = {};

    const int am = tid;                       // A: one 32-elem row per thread
    const int bk = tid >> 3, bn = (tid & 7) * 16;
    const __bf16* Arow = A + (size_t)(m0 + am) * Kp;

    // ---- prologue: stage k0 = 0 into buffer 0 ----
    {
        async_cp16(Arow,      &As[0][am][0]);
        async_cp16(Arow + 8,  &As[0][am][8]);
        async_cp16(Arow + 16, &As[0][am][16]);
        async_cp16(Arow + 24, &As[0][am][24]);
        union { uint4 q[2]; __bf16 e[16]; } tb;
        const uint4* gb = reinterpret_cast<const uint4*>(
            Bm + (size_t)bk * Np + n0 + bn);
        tb.q[0] = gb[0]; tb.q[1] = gb[1];
        #pragma unroll
        for (int j = 0; j < 16; j++) Bs[0][bn + j][bk] = tb.e[j];
        async_wait();
    }
    __syncthreads();

    for (int k0 = 0; k0 < Kp; k0 += 32) {
        const int cur = (k0 >> 5) & 1;
        const bool more = (k0 + 32) < Kp;

        union { uint4 q[2]; __bf16 e[16]; } tb;
        if (more) {
            const __bf16* ga = Arow + k0 + 32;
            async_cp16(ga,      &As[cur ^ 1][am][0]);
            async_cp16(ga + 8,  &As[cur ^ 1][am][8]);
            async_cp16(ga + 16, &As[cur ^ 1][am][16]);
            async_cp16(ga + 24, &As[cur ^ 1][am][24]);
            const uint4* gb = reinterpret_cast<const uint4*>(
                Bm + (size_t)(k0 + 32 + bk) * Np + n0 + bn);
            tb.q[0] = gb[0]; tb.q[1] = gb[1];
        }

        // ---- compute from buffer `cur` ----
        union { uint2 u[4]; v16bf v; } fa[4];
        union { uint4 q[2]; v16bf v; } fb[4];
        #pragma unroll
        for (int mt = 0; mt < 4; mt++) {
            const __bf16* ap = &As[cur][wm * 64 + mt * 16 + l15][half * 8];
            fa[mt].u[0] = *reinterpret_cast<const uint2*>(ap);
            fa[mt].u[1] = *reinterpret_cast<const uint2*>(ap + 4);
            fa[mt].u[2] = *reinterpret_cast<const uint2*>(ap + 16);
            fa[mt].u[3] = *reinterpret_cast<const uint2*>(ap + 20);
        }
        #pragma unroll
        for (int nt = 0; nt < 4; nt++) {
            const __bf16* bp = &Bs[cur][wn * 64 + nt * 16 + l15][half * 8];
            fb[nt].q[0] = *reinterpret_cast<const uint4*>(bp);
            fb[nt].q[1] = *reinterpret_cast<const uint4*>(bp + 16);
        }
        #pragma unroll
        for (int mt = 0; mt < 4; mt++)
            #pragma unroll
            for (int nt = 0; nt < 4; nt++)
                acc[mt][nt] = __builtin_amdgcn_wmma_f32_16x16x32_bf16(
                    false, fa[mt].v, false, fb[nt].v,
                    (short)0, acc[mt][nt], false, false);

        if (more) {
            #pragma unroll
            for (int j = 0; j < 16; j++) Bs[cur ^ 1][bn + j][bk] = tb.e[j];
            async_wait();
            __syncthreads();
        }
    }

    // ---- epilogue ----
    #pragma unroll
    for (int mt = 0; mt < 4; mt++) {
        #pragma unroll
        for (int nt = 0; nt < 4; nt++) {
            const int n_g = n0 + wn * 64 + nt * 16 + l15;
            float bv = 0.f;
            if (EPI == 0 || EPI == 1) bv = (n_g < Ntrue) ? bias[n_g] : 0.f;
            #pragma unroll
            for (int r = 0; r < 8; r++) {
                const int m = m0 + wm * 64 + mt * 16 + r + 8 * half;
                float v = acc[mt][nt][r];
                if (EPI == 0) {
                    v += bv;
                    v = 0.5f * v * (1.f + tanhf(0.7978845608028654f *
                                                (v + 0.044715f * v * v * v)));
                    if (n_g >= Ntrue) v = 0.f;
                    Cb[(size_t)m * Np + n_g] = (__bf16)v;
                } else if (EPI == 1) {
                    v += bv;
                    Cb[(size_t)m * Np + n_g] = (__bf16)v;
                } else if (EPI == 2) {
                    Cb[(size_t)m * Np + n_g] = (__bf16)v;
                } else {
                    int bi  = m / (S_ * TT_);
                    int rem = m % (S_ * TT_);
                    int si = rem / TT_, t = rem % TT_;
                    size_t oidx = ((size_t)bi * L_ + (size_t)t * S_ + si) * C_ + n_g;
                    Cf[oidx] = resid[oidx] + v;
                }
            }
        }
    }
}

// ---------------------------------------------------------------------------
// RMSNorm + RoPE, in place on q/k slices of QKV (bf16). One thread = 32 dims.
// Only first 24 dims rotate (pos=0 for the 4+4 blocks -> identity).
// freq = 256^(-2p/24) = 2^(-2p/3)
// ---------------------------------------------------------------------------
__global__ __launch_bounds__(256)
void rmsrope_kernel(__bf16* __restrict__ qkv,
                    const float* __restrict__ qg, const float* __restrict__ kg)
{
    int idx = blockIdx.x * 256 + threadIdx.x;         // < M_*48
    int row  = idx / 48;
    int rest = idx % 48;
    int qk   = rest / HEADS_;
    int head = rest % HEADS_;
    int t    = row % TT_;

    __bf16* p = qkv + (size_t)row * (3 * C_) + qk * C_ + head * HD_;
    const float* g = qk ? kg : qg;

    float v[HD_]; float ss = 0.f;
    #pragma unroll
    for (int i = 0; i < HD_; i++) { v[i] = (float)p[i]; ss += v[i] * v[i]; }
    float sc = rsqrtf(ss * (1.f / HD_) + 1e-6f);
    #pragma unroll
    for (int i = 0; i < HD_; i++) v[i] *= sc * g[i];

    #pragma unroll
    for (int pr = 0; pr < 12; pr++) {
        float fr  = exp2f(-(float)(2 * pr) * (1.0f / 3.0f));
        float ang = (float)t * fr;
        float c = __cosf(ang), s = __sinf(ang);
        float x0 = v[2 * pr], x1 = v[2 * pr + 1];
        v[2 * pr]     = x0 * c - x1 * s;
        v[2 * pr + 1] = x1 * c + x0 * s;
    }
    #pragma unroll
    for (int i = 0; i < HD_; i++) p[i] = (__bf16)v[i];
}

// ---------------------------------------------------------------------------
// Attention: one wave per (seq, head). scores = 1 WMMA, softmax via width-16
// lane shuffles, P@V = 2 WMMAs (P zero-padded to 16x32, V transposed in LDS).
// ---------------------------------------------------------------------------
__global__ __launch_bounds__(256)
void attn_kernel(const __bf16* __restrict__ qkv, __bf16* __restrict__ attn)
{
    __shared__ __bf16 q_s[8][16][32];
    __shared__ __bf16 k_s[8][16][32];
    __shared__ __bf16 vT[8][32][32];   // [d][m], m=16..31 zero
    __shared__ __bf16 p_s[8][16][32];  // [m][k], k=16..31 zero

    const int tid  = threadIdx.x;
    const int w    = tid >> 5, lane = tid & 31;
    const int half = lane >> 4, l15 = lane & 15;
    const int sh   = blockIdx.x * 8 + w;
    const int seq  = sh / HEADS_, head = sh % HEADS_;
    const size_t rb = (size_t)seq * TT_;

    {
        const int t = lane >> 1, off = (lane & 1) * 16;
        const size_t base = (rb + t) * (3 * C_) + head * HD_ + off;
        const uint4* gq = reinterpret_cast<const uint4*>(qkv + base);
        const uint4* gk = reinterpret_cast<const uint4*>(qkv + base + C_);
        const uint4* gv = reinterpret_cast<const uint4*>(qkv + base + 2 * C_);
        uint4 a0 = gq[0], a1 = gq[1];
        *reinterpret_cast<uint4*>(&q_s[w][t][off])     = a0;
        *reinterpret_cast<uint4*>(&q_s[w][t][off + 8]) = a1;
        uint4 b0 = gk[0], b1 = gk[1];
        *reinterpret_cast<uint4*>(&k_s[w][t][off])     = b0;
        *reinterpret_cast<uint4*>(&k_s[w][t][off + 8]) = b1;
        union { uint4 q[2]; __bf16 e[16]; } tv;
        tv.q[0] = gv[0]; tv.q[1] = gv[1];
        #pragma unroll
        for (int j = 0; j < 16; j++) vT[w][off + j][t] = tv.e[j];
        const uint4 z4 = {0, 0, 0, 0};
        *reinterpret_cast<uint4*>(&vT[w][lane][16]) = z4;
        *reinterpret_cast<uint4*>(&vT[w][lane][24]) = z4;
        if (lane < 16) {
            *reinterpret_cast<uint4*>(&p_s[w][lane][16]) = z4;
            *reinterpret_cast<uint4*>(&p_s[w][lane][24]) = z4;
        }
    }
    __syncthreads();

    union FA { uint2 u[4]; v16bf v; };
    union FB { uint4 q[2]; v16bf v; };
    v8f zero = {};

    FA fq; FB fk;
    {
        const __bf16* ap = &q_s[w][l15][half * 8];
        fq.u[0] = *reinterpret_cast<const uint2*>(ap);
        fq.u[1] = *reinterpret_cast<const uint2*>(ap + 4);
        fq.u[2] = *reinterpret_cast<const uint2*>(ap + 16);
        fq.u[3] = *reinterpret_cast<const uint2*>(ap + 20);
        const __bf16* bp = &k_s[w][l15][half * 8];
        fk.q[0] = *reinterpret_cast<const uint4*>(bp);
        fk.q[1] = *reinterpret_cast<const uint4*>(bp + 16);
    }
    v8f sc = __builtin_amdgcn_wmma_f32_16x16x32_bf16(
        false, fq.v, false, fk.v, (short)0, zero, false, false);

    const float scale = 0.17677669529663687f;  // 1/sqrt(32)
    #pragma unroll
    for (int r = 0; r < 8; r++) {
        float s0 = sc[r] * scale;
        float mx = s0;
        #pragma unroll
        for (int d = 8; d >= 1; d >>= 1) mx = fmaxf(mx, __shfl_xor(mx, d, 16));
        float e = __expf(s0 - mx);
        float sum = e;
        #pragma unroll
        for (int d = 8; d >= 1; d >>= 1) sum += __shfl_xor(sum, d, 16);
        p_s[w][r + 8 * half][l15] = (__bf16)(e / sum);
    }
    __syncthreads();

    FA fp; FB fv0, fv1;
    {
        const __bf16* ap = &p_s[w][l15][half * 8];
        fp.u[0] = *reinterpret_cast<const uint2*>(ap);
        fp.u[1] = *reinterpret_cast<const uint2*>(ap + 4);
        fp.u[2] = *reinterpret_cast<const uint2*>(ap + 16);
        fp.u[3] = *reinterpret_cast<const uint2*>(ap + 20);
        const __bf16* b0 = &vT[w][l15][half * 8];
        fv0.q[0] = *reinterpret_cast<const uint4*>(b0);
        fv0.q[1] = *reinterpret_cast<const uint4*>(b0 + 16);
        const __bf16* b1 = &vT[w][16 + l15][half * 8];
        fv1.q[0] = *reinterpret_cast<const uint4*>(b1);
        fv1.q[1] = *reinterpret_cast<const uint4*>(b1 + 16);
    }
    v8f o0 = __builtin_amdgcn_wmma_f32_16x16x32_bf16(
        false, fp.v, false, fv0.v, (short)0, zero, false, false);
    v8f o1 = __builtin_amdgcn_wmma_f32_16x16x32_bf16(
        false, fp.v, false, fv1.v, (short)0, zero, false, false);

    #pragma unroll
    for (int r = 0; r < 8; r++) {
        size_t orow = rb + r + 8 * half;
        attn[orow * C_ + head * HD_ + l15]      = (__bf16)o0[r];
        attn[orow * C_ + head * HD_ + 16 + l15] = (__bf16)o1[r];
    }
}

// ---------------------------------------------------------------------------
extern "C" void kernel_launch(void* const* d_in, const int* in_sizes, int n_in,
                              void* d_out, int out_size, void* d_ws, size_t ws_size,
                              hipStream_t stream)
{
    (void)in_sizes; (void)n_in; (void)out_size; (void)ws_size;
    const float* mouse = (const float*)d_in[0];
    const float* x     = (const float*)d_in[2];
    const float* w1    = (const float*)d_in[3];
    const float* b1    = (const float*)d_in[4];
    const float* w2    = (const float*)d_in[5];
    const float* b2    = (const float*)d_in[6];
    const float* qkvw  = (const float*)d_in[7];
    const float* qg    = (const float*)d_in[8];
    const float* kg    = (const float*)d_in[9];
    const float* pw    = (const float*)d_in[10];

    char* ws = (char*)d_ws;
    __bf16* Z    = (__bf16*)(ws);                    // 49920*800*2  = 79,872,000
    __bf16* ATTN = Z;                                // reused after gemm1
    __bf16* H    = (__bf16*)(ws + 79872000ull);      // 49920*896*2  = 89,456,640
    __bf16* FEAT = (__bf16*)(ws + 169328640ull);     // 49920*768*2  = 76,677,120
    __bf16* QKV  = (__bf16*)(ws + 246005760ull);     // 49920*2304*2 = 230,031,360
    __bf16* W1s  = (__bf16*)(ws + 476037120ull);     // 800*896*2
    __bf16* W2s  = (__bf16*)(ws + 477470720ull);     // 896*768*2
    __bf16* QWs  = (__bf16*)(ws + 478846976ull);     // 768*2304*2
    __bf16* PWs  = (__bf16*)(ws + 482385920ull);     // 768*768*2

    stage_z_kernel<<<156000, 256, 0, stream>>>(x, mouse, Z);
    stage_w_kernel<<<2800, 256, 0, stream>>>(w1,   W1s, 792, 792, 800, 896);
    stage_w_kernel<<<2688, 256, 0, stream>>>(w2,   W2s, 792, 768, 896, 768);
    stage_w_kernel<<<6912, 256, 0, stream>>>(qkvw, QWs, 768, 2304, 768, 2304);
    stage_w_kernel<<<2304, 256, 0, stream>>>(pw,   PWs, 768, 768, 768, 768);

    gemm_bf16_wmma<0><<<dim3(7, 195),  256, 0, stream>>>(Z,    W1s, b1, H,    nullptr, nullptr, 800, 896, 792);
    gemm_bf16_wmma<1><<<dim3(6, 195),  256, 0, stream>>>(H,    W2s, b2, FEAT, nullptr, nullptr, 896, 768, 768);
    gemm_bf16_wmma<2><<<dim3(18, 195), 256, 0, stream>>>(FEAT, QWs, nullptr, QKV, nullptr, nullptr, 768, 2304, 2304);

    rmsrope_kernel<<<9360, 256, 0, stream>>>(QKV, qg, kg);
    attn_kernel<<<9360, 256, 0, stream>>>(QKV, ATTN);

    gemm_bf16_wmma<3><<<dim3(6, 195), 256, 0, stream>>>(ATTN, PWs, nullptr, nullptr,
                                                        (float*)d_out, x, 768, 768, 768);
}
